// Model_48266842472625
// MI455X (gfx1250) — matrix operand order repaired and checked by reading, not compiled, not run on hardware
//
#include <hip/hip_runtime.h>
#include <hip/hip_bf16.h>

// ---------- sizes ----------
#define HID   256
#define ND    20000
#define NDIS  20000
#define NE    400000
#define NEL   100000
#define NLAYERS 4

// ---------- types ----------
typedef float  v8f   __attribute__((ext_vector_type(8)));
typedef __bf16 v16bf __attribute__((ext_vector_type(16)));
typedef __bf16 v8bf  __attribute__((ext_vector_type(8)));

// fp32 -> bf16 (round to nearest even)
__device__ __forceinline__ unsigned short f2bf(float f) {
    unsigned int u = __float_as_uint(f);
    u += 0x7FFFu + ((u >> 16) & 1u);
    return (unsigned short)(u >> 16);
}
__device__ __forceinline__ float bf2f(unsigned short h) {
    return __uint_as_float(((unsigned int)h) << 16);
}
__device__ __forceinline__ ushort4 f2bf4(float4 v) {
    ushort4 o;
    o.x = f2bf(v.x); o.y = f2bf(v.y); o.z = f2bf(v.z); o.w = f2bf(v.w);
    return o;
}

// ------------------------------------------------------------------
// Pack a row-major fp32 weight W[K][N] into bf16 B-fragment layout:
//   packed[((kt*NT + nt)*32 + lane)*16 + e] = W[kt*32 + (e>>3)*16 + (lane>>4)*8 + (e&7)][nt*16 + (lane&15)]
// so each lane's 16 bf16 (32B) for one 16x16x32 WMMA B tile are contiguous.
// ------------------------------------------------------------------
__global__ void pack_w_kernel(const float* __restrict__ W,
                              unsigned short* __restrict__ out,
                              int K, int N) {
    int t = blockIdx.x * blockDim.x + threadIdx.x;
    if (t >= K * N) return;
    int NT   = N >> 4;
    int e    = t & 15;
    int r    = t >> 4;
    int lane = r & 31; r >>= 5;
    int nt   = r % NT;
    int kt   = r / NT;
    int n    = lane & 15;
    int half = lane >> 4;
    int k    = kt * 32 + (e >> 3) * 16 + half * 8 + (e & 7);
    out[t]   = f2bf(W[(size_t)k * N + nt * 16 + n]);
}

// ------------------------------------------------------------------
// x_dis = disease_x @ lin_w + lin_b + disease_emb[disease_node_id]
// ------------------------------------------------------------------
__global__ void init_xdis_kernel(const float* __restrict__ dx,
                                 const int*   __restrict__ ids,
                                 const float* __restrict__ lw,
                                 const float* __restrict__ lb,
                                 const float* __restrict__ emb,
                                 float* __restrict__ xdis) {
    int t = blockIdx.x * blockDim.x + threadIdx.x;
    if (t >= NDIS * HID) return;
    int n = t >> 8, c = t & 255;
    int id = ids[n];
    float acc = lb[c] + emb[(size_t)id * HID + c];
    const float* row = dx + (size_t)n * 10;
#pragma unroll
    for (int j = 0; j < 10; ++j) acc += row[j] * lw[j * HID + c];
    xdis[t] = acc;
}

__global__ void init_xdrug_kernel(const int* __restrict__ ids,
                                  const float* __restrict__ emb,
                                  float* __restrict__ xd) {
    int t = blockIdx.x * blockDim.x + threadIdx.x;
    if (t >= ND * HID) return;
    int n = t >> 8, c = t & 255;
    xd[t] = emb[(size_t)ids[n] * HID + c];
}

__global__ void zero_kernel(float4* __restrict__ p, int n4) {
    int t = blockIdx.x * blockDim.x + threadIdx.x;
    if (t < n4) p[t] = make_float4(0.f, 0.f, 0.f, 0.f);
}

// ------------------------------------------------------------------
// One wave per edge: gather x[src] (256 f32, 8/lane), scatter-add into
// agg[dst] via native f32 global atomics; lane 0 bumps the edge count.
// ------------------------------------------------------------------
__global__ void scatter_add_kernel(const float* __restrict__ x,
                                   const int* __restrict__ src,
                                   const int* __restrict__ dst,
                                   float* __restrict__ agg,
                                   float* __restrict__ cnt,
                                   int nE) {
    int lane = threadIdx.x & 31;
    int e = (blockIdx.x * blockDim.x + threadIdx.x) >> 5;
    if (e >= nE) return;
    int s = src[e], d = dst[e];
    const float4* xr = reinterpret_cast<const float4*>(x + (size_t)s * HID);
    float4 v0 = xr[lane];
    float4 v1 = xr[lane + 32];
    float* a0 = agg + (size_t)d * HID + lane * 4;
    float* a1 = a0 + 128;
    __hip_atomic_fetch_add(a0 + 0, v0.x, __ATOMIC_RELAXED, __HIP_MEMORY_SCOPE_AGENT);
    __hip_atomic_fetch_add(a0 + 1, v0.y, __ATOMIC_RELAXED, __HIP_MEMORY_SCOPE_AGENT);
    __hip_atomic_fetch_add(a0 + 2, v0.z, __ATOMIC_RELAXED, __HIP_MEMORY_SCOPE_AGENT);
    __hip_atomic_fetch_add(a0 + 3, v0.w, __ATOMIC_RELAXED, __HIP_MEMORY_SCOPE_AGENT);
    __hip_atomic_fetch_add(a1 + 0, v1.x, __ATOMIC_RELAXED, __HIP_MEMORY_SCOPE_AGENT);
    __hip_atomic_fetch_add(a1 + 1, v1.y, __ATOMIC_RELAXED, __HIP_MEMORY_SCOPE_AGENT);
    __hip_atomic_fetch_add(a1 + 2, v1.z, __ATOMIC_RELAXED, __HIP_MEMORY_SCOPE_AGENT);
    __hip_atomic_fetch_add(a1 + 3, v1.w, __ATOMIC_RELAXED, __HIP_MEMORY_SCOPE_AGENT);
    if (lane == 0)
        __hip_atomic_fetch_add(cnt + d, 1.0f, __ATOMIC_RELAXED, __HIP_MEMORY_SCOPE_AGENT);
}

// agg/count -> bf16 (mean), plus bf16 snapshot of layer input x. 4 elems/thread.
__global__ void norm_cvt_kernel(const float4* __restrict__ agg,
                                const float* __restrict__ cnt,
                                const float4* __restrict__ x,
                                ushort4* __restrict__ aggbf,
                                ushort4* __restrict__ xbf,
                                int total4) {
    int t = blockIdx.x * blockDim.x + threadIdx.x;
    if (t >= total4) return;
    float inv = 1.0f / fmaxf(cnt[t >> 6], 1.0f);   // 64 float4 per 256-col row
    float4 a = agg[t];
    a.x *= inv; a.y *= inv; a.z *= inv; a.w *= inv;
    aggbf[t] = f2bf4(a);
    xbf[t]   = f2bf4(x[t]);
}

// ------------------------------------------------------------------
// WMMA GEMM, compile-time shapes: out[M,N] = A[M,K1]@Wp (+ A2[M,K2]@Wp2)
// + bias, optional ReLU, fp32 and/or bf16 output.
// Each wave computes 32 rows x 64 cols: 2 A fragments share every B
// fragment; K loop fully unrolled so B addresses become immediate
// offsets off one base pointer.
// ------------------------------------------------------------------
template <int K1, int K2, int N>
__global__ void gemm_bf16_t(const unsigned short* __restrict__ A,
                            const unsigned short* __restrict__ Wp,
                            const unsigned short* __restrict__ A2,
                            const unsigned short* __restrict__ Wp2,
                            const float* __restrict__ bias,
                            float* __restrict__ outF,
                            unsigned short* __restrict__ outBf,
                            int M, int relu) {
    constexpr int NT  = N / 16;
    constexpr int NCG = N / 64;
    int lane = threadIdx.x & 31;
    int wave = (blockIdx.x * blockDim.x + threadIdx.x) >> 5;
    int rowTile = wave / NCG;          // 32-row tiles
    int cg      = wave % NCG;
    if (rowTile * 32 >= M) return;     // wave-uniform guard (EXEC stays full)
    int mr   = lane & 15;
    int half = lane >> 4;

    v8f acc[2][4] = {};

    // ---- term 1 ----
    {
        const unsigned short* Ar0 = A + (size_t)(rowTile * 32 + mr) * K1 + half * 8;
        const unsigned short* Ar1 = Ar0 + (size_t)16 * K1;
        const unsigned short* Wb  = Wp + (size_t)cg * 4 * 512 + (size_t)lane * 16;
#pragma unroll
        for (int kt = 0; kt < K1 / 32; ++kt) {
            v8bf l0 = *reinterpret_cast<const v8bf*>(Ar0 + kt * 32);
            v8bf h0 = *reinterpret_cast<const v8bf*>(Ar0 + kt * 32 + 16);
            v8bf l1 = *reinterpret_cast<const v8bf*>(Ar1 + kt * 32);
            v8bf h1 = *reinterpret_cast<const v8bf*>(Ar1 + kt * 32 + 16);
            v16bf a0, a1;
#pragma unroll
            for (int i = 0; i < 8; ++i) {
                a0[i] = l0[i]; a0[8 + i] = h0[i];
                a1[i] = l1[i]; a1[8 + i] = h1[i];
            }
#pragma unroll
            for (int j = 0; j < 4; ++j) {
                const v16bf b = *reinterpret_cast<const v16bf*>(
                    Wb + ((size_t)kt * NT + j) * 512);
                acc[0][j] = __builtin_amdgcn_wmma_f32_16x16x32_bf16(
                    false, a0, false, b, (short)0, acc[0][j], false, false);
                acc[1][j] = __builtin_amdgcn_wmma_f32_16x16x32_bf16(
                    false, a1, false, b, (short)0, acc[1][j], false, false);
            }
        }
    }
    // ---- term 2 (SAGE: x @ Wr) ----
    if constexpr (K2 > 0) {
        const unsigned short* Ar0 = A2 + (size_t)(rowTile * 32 + mr) * K2 + half * 8;
        const unsigned short* Ar1 = Ar0 + (size_t)16 * K2;
        const unsigned short* Wb  = Wp2 + (size_t)cg * 4 * 512 + (size_t)lane * 16;
#pragma unroll
        for (int kt = 0; kt < K2 / 32; ++kt) {
            v8bf l0 = *reinterpret_cast<const v8bf*>(Ar0 + kt * 32);
            v8bf h0 = *reinterpret_cast<const v8bf*>(Ar0 + kt * 32 + 16);
            v8bf l1 = *reinterpret_cast<const v8bf*>(Ar1 + kt * 32);
            v8bf h1 = *reinterpret_cast<const v8bf*>(Ar1 + kt * 32 + 16);
            v16bf a0, a1;
#pragma unroll
            for (int i = 0; i < 8; ++i) {
                a0[i] = l0[i]; a0[8 + i] = h0[i];
                a1[i] = l1[i]; a1[8 + i] = h1[i];
            }
#pragma unroll
            for (int j = 0; j < 4; ++j) {
                const v16bf b = *reinterpret_cast<const v16bf*>(
                    Wb + ((size_t)kt * NT + j) * 512);
                acc[0][j] = __builtin_amdgcn_wmma_f32_16x16x32_bf16(
                    false, a0, false, b, (short)0, acc[0][j], false, false);
                acc[1][j] = __builtin_amdgcn_wmma_f32_16x16x32_bf16(
                    false, a1, false, b, (short)0, acc[1][j], false, false);
            }
        }
    }

    // ---- epilogue: C/D layout -> row = tileBase + half*8 + r, col = cg*64 + j*16 + (lane&15)
#pragma unroll
    for (int i = 0; i < 2; ++i) {
        int rowBase = rowTile * 32 + i * 16 + half * 8;
#pragma unroll
        for (int j = 0; j < 4; ++j) {
            int col = cg * 64 + j * 16 + (lane & 15);
            float bv = bias ? bias[col] : 0.f;
#pragma unroll
            for (int r = 0; r < 8; ++r) {
                float v = acc[i][j][r] + bv;
                if (relu) v = fmaxf(v, 0.f);
                size_t o = (size_t)(rowBase + r) * N + col;
                if (outF)  outF[o]  = v;
                if (outBf) outBf[o] = f2bf(v);
            }
        }
    }
}

// build concat([x_drug[i], x_dis[j]]) rows in bf16, 4 cols/thread
__global__ void gather_concat_kernel(const float4* __restrict__ xd,
                                     const float4* __restrict__ xs,
                                     const int* __restrict__ ia,
                                     const int* __restrict__ ib,
                                     ushort4* __restrict__ out,
                                     int nRows) {
    long long t = (long long)blockIdx.x * blockDim.x + threadIdx.x;
    if (t >= (long long)nRows * 128) return;      // 512 cols / 4
    int i = (int)(t >> 7), c4 = (int)(t & 127);
    float4 v = (c4 < 64) ? xd[(size_t)ia[i] * 64 + c4]
                         : xs[(size_t)ib[i] * 64 + (c4 - 64)];
    out[t] = f2bf4(v);
}

// fc4: per-row 64-dot, one wave per row
__global__ void fc4_kernel(const unsigned short* __restrict__ h,
                           const float* __restrict__ w,
                           const float* __restrict__ b,
                           float* __restrict__ out, int M) {
    int lane = threadIdx.x & 31;
    int row = (blockIdx.x * blockDim.x + threadIdx.x) >> 5;
    if (row >= M) return;
    const unsigned short* hr = h + (size_t)row * 64 + lane * 2;
    float s = bf2f(hr[0]) * w[lane * 2] + bf2f(hr[1]) * w[lane * 2 + 1];
#pragma unroll
    for (int off = 16; off; off >>= 1) s += __shfl_xor(s, off, 32);
    if (lane == 0) out[row] = s + b[0];
}

// ------------------------------------------------------------------
extern "C" void kernel_launch(void* const* d_in, const int* in_sizes, int n_in,
                              void* d_out, int out_size, void* d_ws, size_t ws_size,
                              hipStream_t stream) {
    const int*   drug_id  = (const int*)  d_in[0];
    const float* dis_x    = (const float*)d_in[1];
    const int*   dis_id   = (const int*)  d_in[2];
    const int*   e_mt     = (const int*)  d_in[3];   // [2,E]
    const int*   e_rev    = (const int*)  d_in[4];
    const int*   e_lab    = (const int*)  d_in[5];   // [2,EL]
    const float* drug_emb = (const float*)d_in[6];
    const float* dis_emb  = (const float*)d_in[7];
    const float* lin_w    = (const float*)d_in[8];
    const float* lin_b    = (const float*)d_in[9];
    const float* Wl_mt    = (const float*)d_in[10];
    const float* bl_mt    = (const float*)d_in[11];
    const float* Wr_mt    = (const float*)d_in[12];
    const float* Wl_rev   = (const float*)d_in[13];
    const float* bl_rev   = (const float*)d_in[14];
    const float* Wr_rev   = (const float*)d_in[15];
    const float* fc1_w = (const float*)d_in[16]; const float* fc1_b = (const float*)d_in[17];
    const float* fc2_w = (const float*)d_in[18]; const float* fc2_b = (const float*)d_in[19];
    const float* fc3_w = (const float*)d_in[20]; const float* fc3_b = (const float*)d_in[21];
    const float* fc4_w = (const float*)d_in[22]; const float* fc4_b = (const float*)d_in[23];

    // ---- workspace layout ----
    char* ws = (char*)d_ws;
    size_t off = 0;
    auto alloc = [&](size_t bytes) {
        void* p = ws + off;
        off = (off + bytes + 255) & ~(size_t)255;
        return p;
    };
    const size_t WSZ = (size_t)HID * HID * 2;                    // packed 256x256 bf16
    unsigned short* convp = (unsigned short*)alloc(16 * WSZ);    // [WlMt x4][WrMt x4][WlRev x4][WrRev x4]
    unsigned short* fc1p  = (unsigned short*)alloc(512 * 256 * 2);
    unsigned short* fc2p  = (unsigned short*)alloc(256 * 128 * 2);
    unsigned short* fc3p  = (unsigned short*)alloc(128 * 64 * 2);
    float* x_dis  = (float*)alloc((size_t)NDIS * HID * 4);
    float* x_drug = (float*)alloc((size_t)ND   * HID * 4);
    size_t mlpStart = off;                                       // aliased region begins here
    float* agg_dis  = (float*)alloc((size_t)NDIS * HID * 4);
    float* agg_drug = (float*)alloc((size_t)ND   * HID * 4);
    float* cnt_dis  = (float*)alloc((size_t)NDIS * 4);
    float* cnt_drug = (float*)alloc((size_t)ND   * 4);
    unsigned short* aggdis_bf  = (unsigned short*)alloc((size_t)NDIS * HID * 2);
    unsigned short* xdis_bf    = (unsigned short*)alloc((size_t)NDIS * HID * 2);
    unsigned short* aggdrug_bf = (unsigned short*)alloc((size_t)ND * HID * 2);
    unsigned short* xdrug_bf   = (unsigned short*)alloc((size_t)ND * HID * 2);
    // MLP scratch reuses the (dead) aggregation region
    off = mlpStart;
    unsigned short* ebf = (unsigned short*)alloc((size_t)NEL * 512 * 2);
    unsigned short* h1  = (unsigned short*)alloc((size_t)NEL * 256 * 2);
    unsigned short* h2  = (unsigned short*)alloc((size_t)NEL * 128 * 2);
    unsigned short* h3  = (unsigned short*)alloc((size_t)NEL * 64 * 2);
    (void)ws_size; (void)in_sizes; (void)n_in; (void)out_size;

    auto nb = [](long long threads) { return (unsigned)((threads + 255) / 256); };

    // ---- pack weights (bf16, WMMA B-fragment order) ----
    for (int l = 0; l < NLAYERS; ++l) {
        pack_w_kernel<<<nb(HID * HID), 256, 0, stream>>>(Wl_mt  + (size_t)l * HID * HID, convp + (0 * 4 + l) * (WSZ / 2), HID, HID);
        pack_w_kernel<<<nb(HID * HID), 256, 0, stream>>>(Wr_mt  + (size_t)l * HID * HID, convp + (1 * 4 + l) * (WSZ / 2), HID, HID);
        pack_w_kernel<<<nb(HID * HID), 256, 0, stream>>>(Wl_rev + (size_t)l * HID * HID, convp + (2 * 4 + l) * (WSZ / 2), HID, HID);
        pack_w_kernel<<<nb(HID * HID), 256, 0, stream>>>(Wr_rev + (size_t)l * HID * HID, convp + (3 * 4 + l) * (WSZ / 2), HID, HID);
    }
    pack_w_kernel<<<nb(512 * 256), 256, 0, stream>>>(fc1_w, fc1p, 512, 256);
    pack_w_kernel<<<nb(256 * 128), 256, 0, stream>>>(fc2_w, fc2p, 256, 128);
    pack_w_kernel<<<nb(128 * 64),  256, 0, stream>>>(fc3_w, fc3p, 128, 64);

    // ---- embeddings ----
    init_xdis_kernel<<<nb((long long)NDIS * HID), 256, 0, stream>>>(dis_x, dis_id, lin_w, lin_b, dis_emb, x_dis);
    init_xdrug_kernel<<<nb((long long)ND * HID), 256, 0, stream>>>(drug_id, drug_emb, x_drug);

    // ---- SAGE layers ----
    // zero span covers agg_dis..agg_drug..cnt_dis..cnt_drug incl. alignment pads
    const size_t zeroBytes = (size_t)((char*)cnt_drug - (char*)agg_dis) + (size_t)ND * 4;
    const int zeroN4 = (int)(zeroBytes / 16);
    for (int l = 0; l < NLAYERS; ++l) {
        zero_kernel<<<nb(zeroN4), 256, 0, stream>>>((float4*)agg_dis, zeroN4);
        scatter_add_kernel<<<nb((long long)NE * 32), 256, 0, stream>>>(x_drug, e_mt, e_mt + NE, agg_dis, cnt_dis, NE);
        scatter_add_kernel<<<nb((long long)NE * 32), 256, 0, stream>>>(x_dis, e_rev, e_rev + NE, agg_drug, cnt_drug, NE);
        norm_cvt_kernel<<<nb((long long)NDIS * HID / 4), 256, 0, stream>>>(
            (const float4*)agg_dis, cnt_dis, (const float4*)x_dis,
            (ushort4*)aggdis_bf, (ushort4*)xdis_bf, NDIS * HID / 4);
        norm_cvt_kernel<<<nb((long long)ND * HID / 4), 256, 0, stream>>>(
            (const float4*)agg_drug, cnt_drug, (const float4*)x_drug,
            (ushort4*)aggdrug_bf, (ushort4*)xdrug_bf, ND * HID / 4);
        int relu = (l < NLAYERS - 1) ? 1 : 0;
        long long waves = (long long)(NDIS / 32) * (HID / 64);
        gemm_bf16_t<256, 256, 256><<<nb(waves * 32), 256, 0, stream>>>(
            aggdis_bf, convp + (0 * 4 + l) * (WSZ / 2), xdis_bf, convp + (1 * 4 + l) * (WSZ / 2),
            bl_mt + (size_t)l * HID, x_dis, nullptr, NDIS, relu);
        gemm_bf16_t<256, 256, 256><<<nb(waves * 32), 256, 0, stream>>>(
            aggdrug_bf, convp + (2 * 4 + l) * (WSZ / 2), xdrug_bf, convp + (3 * 4 + l) * (WSZ / 2),
            bl_rev + (size_t)l * HID, x_drug, nullptr, ND, relu);
    }

    // ---- link MLP ----
    gather_concat_kernel<<<nb((long long)NEL * 128), 256, 0, stream>>>(
        (const float4*)x_drug, (const float4*)x_dis, e_lab, e_lab + NEL, (ushort4*)ebf, NEL);
    long long w1 = (long long)(NEL / 32) * (256 / 64);
    gemm_bf16_t<512, 0, 256><<<nb(w1 * 32), 256, 0, stream>>>(
        ebf, fc1p, nullptr, nullptr, fc1_b, nullptr, h1, NEL, 1);
    long long w2 = (long long)(NEL / 32) * (128 / 64);
    gemm_bf16_t<256, 0, 128><<<nb(w2 * 32), 256, 0, stream>>>(
        h1, fc2p, nullptr, nullptr, fc2_b, nullptr, h2, NEL, 1);
    long long w3 = (long long)(NEL / 32) * (64 / 64);
    gemm_bf16_t<128, 0, 64><<<nb(w3 * 32), 256, 0, stream>>>(
        h2, fc3p, nullptr, nullptr, fc3_b, nullptr, h3, NEL, 1);
    fc4_kernel<<<nb((long long)NEL * 32), 256, 0, stream>>>(h3, fc4_w, fc4_b, (float*)d_out, NEL);
}
